// BiLSTM_CRF_69458211111402
// MI455X (gfx1250) — matrix operand order repaired and checked by reading, not compiled, not run on hardware
//
#include <hip/hip_runtime.h>

typedef __bf16 bf16;
typedef __attribute__((ext_vector_type(16))) __bf16 v16bf;
typedef __attribute__((ext_vector_type(8)))  float  v8f;

#define BATCH 64
#define TLEN  512
#define EDIM  512
#define HDIM  512
#define G4    2048   // 4*H
#define H2    1024
#define NTAG  7
#define NPAD  16     // NTAG padded to one WMMA tile

// ---------------- WMMA fragment loaders (contiguous k order: lane half h holds k 16h..16h+15) ----

// A: 16x32 bf16 tile of row-major matrix A[lda], rows row0..row0+15, k0..k0+31.
// Works for global or LDS-resident A (compiler picks global_load / ds_load).
__device__ __forceinline__ v16bf load_a_frag(const bf16* __restrict__ A, int lda,
                                             int row0, int k0, int lane) {
  const bf16* p = A + (size_t)(row0 + (lane & 15)) * lda + k0;
  const int hi = (lane >> 4) * 16;  // lanes 16-31 hold K+16..K+31
  v16bf a;
#pragma unroll
  for (int j = 0; j < 16; ++j) a[j] = p[hi + j];
  return a;
}

// B: 32x16 bf16 tile where B(k,n) = W[n][k], W row-major with stride ldw.
// 16 contiguous bf16 per lane -> vectorizes to global_load_b128 pairs.
__device__ __forceinline__ v16bf load_bT_frag(const bf16* __restrict__ W, int ldw,
                                              int n0, int k0, int lane) {
  const bf16* p = W + (size_t)(n0 + (lane & 15)) * ldw + k0 + (lane >> 4) * 16;
  v16bf b;
#pragma unroll
  for (int j = 0; j < 16; ++j) b[j] = p[j];
  return b;
}

// ---------------- small prep kernels ------------------------------------

__global__ void cvt_f32_bf16(const float* __restrict__ src, bf16* __restrict__ dst, int n) {
  int i = blockIdx.x * 256 + threadIdx.x;
  if (i < n) dst[i] = (bf16)src[i];
}

__global__ void vadd_f32(const float* __restrict__ a, const float* __restrict__ b,
                         float* __restrict__ c, int n) {
  int i = blockIdx.x * 256 + threadIdx.x;
  if (i < n) c[i] = a[i] + b[i];
}

// w_out (7x1024) -> bf16 padded to 16x1024 (rows 7..15 zero)
__global__ void cvt_wout(const float* __restrict__ src, bf16* __restrict__ dst) {
  int i = blockIdx.x * 256 + threadIdx.x;   // over 16*1024
  int tag = i >> 10;
  dst[i] = (tag < NTAG) ? (bf16)src[i] : (bf16)0.0f;
}

// embedding gather + bf16 convert: x[m][e] = emb[sent[m]][e], m = b*T + t
__global__ void embed_kernel(const int* __restrict__ sent, const float* __restrict__ emb,
                             bf16* __restrict__ x) {
  size_t idx = (size_t)blockIdx.x * 256 + threadIdx.x;   // over 32768*512
  int m = (int)(idx >> 9);
  int e = (int)(idx & 511);
  x[idx] = (bf16)emb[(size_t)sent[m] * EDIM + e];
}

// ---------------- bulk GEMM: input projections ---------------------------
// pre[t][b][n] = bias[n] + sum_k x[b*T+t][k] * w_ih[n][k]
// M = B*T = 32768, N = 2048, K = 512.  Block = 8 waves; tile 128M x 64N.
// Output written in [t][B][4H] order so each LSTM step reads one dense block.
__global__ __launch_bounds__(256) void gemm_pre(const bf16* __restrict__ A,
                                                const bf16* __restrict__ W,
                                                const float* __restrict__ bias,
                                                bf16* __restrict__ out) {
  const int lane = threadIdx.x & 31;
  const int wave = threadIdx.x >> 5;
  const int m0 = blockIdx.y * 128 + wave * 16;
  const int nb = blockIdx.x * 64;
  v8f acc[4] = {};
  for (int k0 = 0; k0 < EDIM; k0 += 32) {
    v16bf a = load_a_frag(A, EDIM, m0, k0, lane);
#pragma unroll
    for (int nt = 0; nt < 4; ++nt) {
      v16bf b = load_bT_frag(W, EDIM, nb + nt * 16, k0, lane);
      acc[nt] = __builtin_amdgcn_wmma_f32_16x16x32_bf16(false, a, false, b,
                                                        (short)0, acc[nt], false, false);
    }
  }
  const int col_l = lane & 15;
  const int rbase = (lane >> 4) * 8;
#pragma unroll
  for (int nt = 0; nt < 4; ++nt) {
    int col = nb + nt * 16 + col_l;
    float bv = bias[col];
#pragma unroll
    for (int r = 0; r < 8; ++r) {
      int row = m0 + rbase + r;            // m = b*T + t
      int tt = row & (TLEN - 1);
      int bb = row >> 9;
      out[((size_t)tt * BATCH + bb) * G4 + col] = (bf16)(acc[nt][r] + bv);
    }
  }
}

// ---------------- fused per-timestep LSTM step (fwd+bwd in one launch) ---
// grid = (16, 2): y = direction. Block owns 32 hidden units => 128 gate cols
// {i,f,g,o}; 8 waves: wave w -> gate (w>>1), 16-col subtile; 4 M-tiles (B=64).
// h_in (64KB) is staged in LDS once per block; after the K-loop the same LDS
// is re-used as the 32KB f32 gate-exchange tile (aliased, barrier-separated).
__global__ __launch_bounds__(256) void lstm_step_dual(
    const bf16* __restrict__ whhf, const bf16* __restrict__ whhb,
    const bf16* __restrict__ pref, const bf16* __restrict__ preb,
    const bf16* __restrict__ hf_in, bf16* __restrict__ hf_out,
    const bf16* __restrict__ hb_in, bf16* __restrict__ hb_out,
    float* __restrict__ cf, float* __restrict__ cb,
    bf16* __restrict__ lout, int s) {
  __shared__ __align__(16) char smem[BATCH * HDIM * sizeof(bf16)];   // 64 KB
  const int dir = blockIdx.y;
  const bf16* whh  = dir ? whhb  : whhf;
  const bf16* pre  = dir ? preb  : pref;
  const bf16* h_in = dir ? hb_in : hf_in;
  bf16*  h_out = dir ? hb_out : hf_out;
  float* c     = dir ? cb : cf;
  const int t    = dir ? (TLEN - 1 - s) : s;
  const int col0 = dir ? HDIM : 0;

  // stage h_in -> LDS (4096 x b128)
  {
    const uint4* src = (const uint4*)h_in;
    uint4* dst = (uint4*)smem;
    for (int i = threadIdx.x; i < BATCH * HDIM * 2 / 16; i += 256) dst[i] = src[i];
  }
  __syncthreads();
  const bf16* hsm = (const bf16*)smem;

  const int lane = threadIdx.x & 31;
  const int wave = threadIdx.x >> 5;
  const int j0 = blockIdx.x * 32;
  const int gate = wave >> 1;
  const int nG = gate * HDIM + j0 + (wave & 1) * 16;   // global gate-column base
  v8f acc[4] = {};
  for (int k0 = 0; k0 < HDIM; k0 += 32) {
    v16bf b = load_bT_frag(whh, HDIM, nG, k0, lane);
#pragma unroll
    for (int mt = 0; mt < 4; ++mt) {
      v16bf a = load_a_frag(hsm, HDIM, mt * 16, k0, lane);   // ds_load path
      acc[mt] = __builtin_amdgcn_wmma_f32_16x16x32_bf16(false, a, false, b,
                                                        (short)0, acc[mt], false, false);
    }
  }
  __syncthreads();                      // everyone done reading hsm
  float (*g)[128] = (float (*)[128])smem;

  const int col_l = lane & 15;
  const int rbase = (lane >> 4) * 8;
  const int lcol = wave * 16 + col_l;
  const int gcol = nG + col_l;
  const bf16* pre_t = pre + (size_t)t * BATCH * G4;   // dense 64x2048 block
#pragma unroll
  for (int mt = 0; mt < 4; ++mt) {
#pragma unroll
    for (int r = 0; r < 8; ++r) {
      int b = mt * 16 + rbase + r;
      g[b][lcol] = acc[mt][r] + (float)pre_t[(size_t)b * G4 + gcol];
    }
  }
  __syncthreads();
  for (int e = threadIdx.x; e < BATCH * 32; e += 256) {
    int b = e & 63, jl = e >> 6;
    float iv = g[b][jl], fv = g[b][32 + jl], gv = g[b][64 + jl], ov = g[b][96 + jl];
    int j = j0 + jl;
    float cold = c[b * HDIM + j];
    float si = 1.f / (1.f + expf(-iv));
    float sf = 1.f / (1.f + expf(-fv));
    float so = 1.f / (1.f + expf(-ov));
    float cn = sf * cold + si * tanhf(gv);
    float hn = so * tanhf(cn);
    c[b * HDIM + j] = cn;
    h_out[b * HDIM + j] = (bf16)hn;
    lout[((size_t)b * TLEN + t) * H2 + col0 + j] = (bf16)hn;
  }
}

// ---------------- emission logits via WMMA (NTAG padded to 16) -----------
// logits[p][tag] = b_out[tag] + sum_j lout[p][j] * w_out[tag][j]
// Block = 8 waves, each wave: 16 rows x 16 tags, K = 1024 (32 k-steps).
__global__ __launch_bounds__(256) void logits_wmma(const bf16* __restrict__ lout,
                                                   const bf16* __restrict__ woutT,
                                                   const float* __restrict__ b_out,
                                                   float* __restrict__ logits) {
  const int lane = threadIdx.x & 31;
  const int wave = threadIdx.x >> 5;
  const int m0 = blockIdx.x * 128 + wave * 16;
  v8f acc = {};
  for (int k0 = 0; k0 < H2; k0 += 32) {
    v16bf a = load_a_frag(lout, H2, m0, k0, lane);
    v16bf b = load_bT_frag(woutT, H2, 0, k0, lane);
    acc = __builtin_amdgcn_wmma_f32_16x16x32_bf16(false, a, false, b,
                                                  (short)0, acc, false, false);
  }
  const int col = lane & 15;
  const int rbase = (lane >> 4) * 8;
  if (col < NTAG) {
    float bv = b_out[col];
#pragma unroll
    for (int r = 0; r < 8; ++r)
      logits[(size_t)(m0 + rbase + r) * NTAG + col] = acc[r] + bv;
  }
}

// ---------------- Viterbi: one block per batch row, BPs live in LDS ------
__global__ void viterbi_kernel(const float* __restrict__ logits, const float* __restrict__ trans,
                               float* __restrict__ out) {
  __shared__ float delta[8], ndelta[8];
  __shared__ unsigned char bp[TLEN][8];
  const int b = blockIdx.x, tid = threadIdx.x;
  if (tid < NTAG) delta[tid] = logits[((size_t)b * TLEN + 0) * NTAG + tid];
  __syncthreads();
  for (int t = 1; t < TLEN; ++t) {
    if (tid < NTAG) {
      float best = -3.4e38f; int bi = 0;
      for (int i = 0; i < NTAG; ++i) {
        float v = delta[i] + trans[i * NTAG + tid];
        if (v > best) { best = v; bi = i; }
      }
      ndelta[tid] = best + logits[((size_t)b * TLEN + t) * NTAG + tid];
      bp[t][tid] = (unsigned char)bi;
    }
    __syncthreads();
    if (tid < NTAG) delta[tid] = ndelta[tid];
    __syncthreads();
  }
  if (tid == 0) {
    float best = -3.4e38f; int tag = 0;
    for (int j = 0; j < NTAG; ++j) if (delta[j] > best) { best = delta[j]; tag = j; }
    out[b] = best;
    float* paths = out + BATCH;
    paths[(size_t)b * TLEN + TLEN - 1] = (float)tag;
    for (int t = TLEN - 1; t >= 1; --t) {
      tag = bp[t][tag];
      paths[(size_t)b * TLEN + t - 1] = (float)tag;
    }
  }
}

// ---------------- host orchestration -------------------------------------
extern "C" void kernel_launch(void* const* d_in, const int* in_sizes, int n_in,
                              void* d_out, int out_size, void* d_ws, size_t ws_size,
                              hipStream_t stream) {
  const int*   sent  = (const int*)  d_in[0];
  const float* emb   = (const float*)d_in[1];
  const float* wihf  = (const float*)d_in[2];
  const float* whhf  = (const float*)d_in[3];
  const float* bihf  = (const float*)d_in[4];
  const float* bhhf  = (const float*)d_in[5];
  const float* wihb  = (const float*)d_in[6];
  const float* whhb  = (const float*)d_in[7];
  const float* bihb  = (const float*)d_in[8];
  const float* bhhb  = (const float*)d_in[9];
  const float* wout  = (const float*)d_in[10];
  const float* bout  = (const float*)d_in[11];
  const float* h0    = (const float*)d_in[12];
  const float* c0    = (const float*)d_in[13];
  const float* trans = (const float*)d_in[14];
  float* out = (float*)d_out;

  char* base = (char*)d_ws;
  size_t off = 0;
  auto take = [&](size_t bytes) -> void* {
    void* p = base + off;
    off += (bytes + 255) & ~(size_t)255;
    return p;
  };
  const size_t WB = (size_t)G4 * HDIM;          // weight elements
  const size_t M  = (size_t)BATCH * TLEN;       // 32768
  bf16*  wihf_bf = (bf16*)take(WB * 2);
  bf16*  whhf_bf = (bf16*)take(WB * 2);
  bf16*  wihb_bf = (bf16*)take(WB * 2);
  bf16*  whhb_bf = (bf16*)take(WB * 2);
  float* biasf   = (float*)take(G4 * 4);
  float* biasb   = (float*)take(G4 * 4);
  bf16*  woutT   = (bf16*)take((size_t)NPAD * H2 * 2);
  bf16*  xbf     = (bf16*)take(M * EDIM * 2);
  bf16*  pref    = (bf16*)take(M * G4 * 2);     // [t][B][4H]
  bf16*  preb    = (bf16*)take(M * G4 * 2);     // [t][B][4H]
  const size_t HS = (size_t)BATCH * HDIM;       // h buffer stride (elements)
  bf16*  hf      = (bf16*)take(2 * HS * 2);     // ping-pong fwd h
  bf16*  hb      = (bf16*)take(2 * HS * 2);     // ping-pong bwd h
  float* cf      = (float*)take(HS * 4);
  float* cb      = (float*)take(HS * 4);
  bf16*  lout    = (bf16*)take(M * H2 * 2);
  float* logits  = (float*)take(M * NTAG * 4);

  // 1) weight/bias/state prep
  cvt_f32_bf16<<<(int)(WB / 256), 256, 0, stream>>>(wihf, wihf_bf, (int)WB);
  cvt_f32_bf16<<<(int)(WB / 256), 256, 0, stream>>>(whhf, whhf_bf, (int)WB);
  cvt_f32_bf16<<<(int)(WB / 256), 256, 0, stream>>>(wihb, wihb_bf, (int)WB);
  cvt_f32_bf16<<<(int)(WB / 256), 256, 0, stream>>>(whhb, whhb_bf, (int)WB);
  vadd_f32<<<G4 / 256, 256, 0, stream>>>(bihf, bhhf, biasf, G4);
  vadd_f32<<<G4 / 256, 256, 0, stream>>>(bihb, bhhb, biasb, G4);
  cvt_wout<<<NPAD * H2 / 256, 256, 0, stream>>>(wout, woutT);
  cvt_f32_bf16<<<(int)(HS / 256), 256, 0, stream>>>(h0, hf, (int)HS);          // h0[0]
  cvt_f32_bf16<<<(int)(HS / 256), 256, 0, stream>>>(h0 + HS, hb, (int)HS);     // h0[1]
  hipMemcpyAsync(cf, c0, HS * 4, hipMemcpyDeviceToDevice, stream);
  hipMemcpyAsync(cb, c0 + HS, HS * 4, hipMemcpyDeviceToDevice, stream);

  // 2) embedding gather -> bf16
  embed_kernel<<<(int)(M * EDIM / 256), 256, 0, stream>>>(sent, emb, xbf);

  // 3) bulk input-projection GEMMs (WMMA bf16 -> f32)
  gemm_pre<<<dim3(G4 / 64, (int)(M / 128)), 256, 0, stream>>>(xbf, wihf_bf, biasf, pref);
  gemm_pre<<<dim3(G4 / 64, (int)(M / 128)), 256, 0, stream>>>(xbf, wihb_bf, biasb, preb);

  // 4) sequential recurrence: fwd & bwd chains in the same per-step launch
  for (int s = 0; s < TLEN; ++s) {
    bf16* fin  = hf + (size_t)(s & 1) * HS;
    bf16* fout = hf + (size_t)((s + 1) & 1) * HS;
    bf16* bin  = hb + (size_t)(s & 1) * HS;
    bf16* bo   = hb + (size_t)((s + 1) & 1) * HS;
    lstm_step_dual<<<dim3(HDIM / 32, 2), 256, 0, stream>>>(
        whhf_bf, whhb_bf, pref, preb, fin, fout, bin, bo, cf, cb, lout, s);
  }

  // 5) emission logits (WMMA), 6) Viterbi decode (scores || paths)
  logits_wmma<<<(int)(M / 128), 256, 0, stream>>>(lout, woutT, bout, logits);
  viterbi_kernel<<<BATCH, 32, 0, stream>>>(logits, trans, out);
}
